// SelfAttention_57939108823285
// MI455X (gfx1250) — matrix operand order repaired and checked by reading, not compiled, not run on hardware
//
#include <hip/hip_runtime.h>

typedef _Float16 h16;
typedef __attribute__((ext_vector_type(16))) _Float16 v16h;
typedef __attribute__((ext_vector_type(8)))  _Float16 v8h;
typedef __attribute__((ext_vector_type(8)))  float    v8f;

constexpr int NB = 8;     // batch
constexpr int NC = 2048;  // context length
constexpr int ND = 128;   // feature dim
constexpr int NH = 128;   // head dim
constexpr float NEG_SLOPE = 0.01f;

// ---- WMMA fragment helpers (contiguous k order, same for A and B, wave32) ----
// A (16x32 f16): lane m = L&15, hi = L>>4.
//   elems 0..7  = K = kk + hi*16 + {0..7}       (16B contiguous)
//   elems 8..15 = K = kk + hi*16 + 8 + {0..7}   (16B contiguous)
__device__ __forceinline__ v16h ld_a_frag(const h16* row, int kk, int hi) {
  v8h lo = *(const v8h*)(row + kk + hi * 16);
  v8h hv = *(const v8h*)(row + kk + hi * 16 + 8);
  v16h a;
#pragma unroll
  for (int e = 0; e < 8; ++e) { a[e] = lo[e]; a[8 + e] = hv[e]; }
  return a;
}

// B (32x16 f16): lane n = L&15, elems 0..15 = K = kk + hi*16 + {0..15} (32B contiguous)
__device__ __forceinline__ v8f wmma_f16(v16h a, v16h b, v8f c) {
  return __builtin_amdgcn_wmma_f32_16x16x32_f16(false, a, false, b, (short)0, c,
                                                false, false);
}

// ---------------------------------------------------------------------
// Kernel 1: fold score_w through Wq/Wk, convert Wv and out_w to f16.
// ---------------------------------------------------------------------
__global__ __launch_bounds__(256) void prep_kernel(
    const float* __restrict__ w_w, const float* __restrict__ w_b,
    const float* __restrict__ score_w, const float* __restrict__ score_b,
    const float* __restrict__ out_w,
    float* __restrict__ vq, float* __restrict__ vk, float* __restrict__ bqk,
    h16* __restrict__ wv_h, h16* __restrict__ outw_h) {
  const int t = threadIdx.x;
  if (t < ND) {
    float aq = 0.f, ak = 0.f;
    for (int h = 0; h < NH; ++h) {
      aq += score_w[h] * w_w[(size_t)h * ND + t];
      ak += score_w[NH + h] * w_w[(size_t)(NH + h) * ND + t];
    }
    vq[t] = aq;
    vk[t] = ak;
  }
  if (t == 0) {
    float bq = score_b[0], bk = 0.f;
    for (int h = 0; h < NH; ++h) {
      bq += score_w[h] * w_b[h];
      bk += score_w[NH + h] * w_b[NH + h];
    }
    bqk[0] = bq;
    bqk[1] = bk;
  }
  for (int e = t; e < NH * ND; e += 256) {
    wv_h[e] = (h16)w_w[2 * NH * ND + e];  // Wv rows, row-major [h][d]
    outw_h[e] = (h16)out_w[e];            // out_w row-major [d][h]
  }
}

// ---------------------------------------------------------------------
// Kernel 2: per 16 context rows: sq/sk scalars + v projection via WMMA,
// v stored transposed as vT[b][h][c] in f16 for the attention B-fragments.
// ---------------------------------------------------------------------
__global__ __launch_bounds__(256) void qkv_kernel(
    const float* __restrict__ features, const float* __restrict__ w_b,
    const float* __restrict__ vq, const float* __restrict__ vk,
    const float* __restrict__ bqk, const h16* __restrict__ wv_h,
    float* __restrict__ sq, float* __restrict__ sk, h16* __restrict__ vT) {
  __shared__ float sF[16][128];
  __shared__ h16 sFh[16][136];                       // pad: 272B rows, 16B aligned
  __shared__ __align__(32) h16 sStage[8][16][16];    // per-wave transpose stage

  const int t = threadIdx.x;
  const int b = blockIdx.x >> 7;
  const int c0 = (blockIdx.x & 127) << 4;

#pragma unroll
  for (int s = 0; s < 8; ++s) {
    int e = t + (s << 8);
    int i = e >> 7, d = e & 127;
    float f = features[((size_t)b * NC + c0 + i) * ND + d];
    sF[i][d] = f;
    sFh[i][d] = (h16)f;
  }
  __syncthreads();

  if (t < 16) {  // folded q/k scores: one dot of length 128 each
    float dq = 0.f, dk = 0.f;
    for (int d = 0; d < ND; ++d) {
      float f = sF[t][d];
      dq += f * vq[d];
      dk += f * vk[d];
    }
    sq[(size_t)b * NC + c0 + t] = dq + bqk[0];
    sk[(size_t)b * NC + c0 + t] = dk + bqk[1];
  }

  const int lane = t & 31, wave = t >> 5;
  const int hi = lane >> 4, ln = lane & 15;
  const int n0 = wave << 4;

  v8f acc = {};
  const h16* arow = &sFh[ln][0];
  const h16* brow = wv_h + (size_t)(n0 + ln) * ND;
#pragma unroll
  for (int kk = 0; kk < ND; kk += 32) {
    v16h a = ld_a_frag(arow, kk, hi);
    v16h bf = *(const v16h*)(brow + kk + hi * 16);
    acc = wmma_f16(a, bf, acc);
  }
  const float bias = w_b[2 * NH + n0 + ln];
#pragma unroll
  for (int r = 0; r < 8; ++r)
    sStage[wave][r + hi * 8][ln] = (h16)(acc[r] + bias);
  __syncthreads();
  if (lane < 16) {  // coalesced 32B stores of the transposed tile
    v16h row = *(const v16h*)&sStage[wave][lane][0];
    *(v16h*)(vT + ((size_t)b * NH + n0 + lane) * NC + c0) = row;
  }
}

// ---------------------------------------------------------------------
// Kernel 3: per 16 query rows: online softmax stats (pass 1 over mask),
// then 4 chunks of 512 keys: materialize f16 attn in LDS and WMMA against
// vT; finally project weighted (16x128) through out_w with 4 more WMMAs.
// ---------------------------------------------------------------------
__global__ __launch_bounds__(256) void attn_kernel(
    const int* __restrict__ mask, const float* __restrict__ sq,
    const float* __restrict__ sk, const h16* __restrict__ vT,
    const h16* __restrict__ outw_h, const float* __restrict__ out_b,
    float* __restrict__ out) {
  __shared__ h16 sAttn[16][520];   // 1040B rows: 16B aligned, 4-bank skew
  __shared__ float redM[16][16];
  __shared__ float redS[16][16];
  __shared__ float sMax[16];
  __shared__ float sInv[16];
  __shared__ __align__(32) h16 sWh[16][136];

  const int t = threadIdx.x;
  const int b = blockIdx.x >> 7;
  const int i0 = (blockIdx.x & 127) << 4;
  const int i = t >> 4, g = t & 15;

  const float sqv = sq[(size_t)b * NC + i0 + i];
  const float* skb = sk + (size_t)b * NC;
  const int* mrow = mask + ((size_t)b * NC + (i0 + i)) * NC;

  // ---- pass 1: online row max / sumexp (16 threads per row) ----
  float m = -1e30f, s = 0.f;
  for (int it = 0; it < NC / 16; ++it) {
    int j = g + (it << 4);
    float sc = sqv + skb[j];
    sc = sc > 0.f ? sc : NEG_SLOPE * sc;
    sc *= (float)mrow[j];
    float nm = fmaxf(m, sc);
    s = s * __expf(m - nm) + __expf(sc - nm);
    m = nm;
  }
  redM[i][g] = m;
  redS[i][g] = s;
  __syncthreads();
  if (t < 16) {
    float mm = -1e30f;
#pragma unroll
    for (int g2 = 0; g2 < 16; ++g2) mm = fmaxf(mm, redM[t][g2]);
    float ss = 0.f;
#pragma unroll
    for (int g2 = 0; g2 < 16; ++g2) ss += redS[t][g2] * __expf(redM[t][g2] - mm);
    sMax[t] = mm;
    sInv[t] = 1.f / ss;
  }
  __syncthreads();

  const int lane = t & 31, wave = t >> 5;
  const int hi = lane >> 4, ln = lane & 15;
  const int n0 = wave << 4;

  const float mi = sMax[i];
  const float inv = sInv[i];
  const h16* brow0 = vT + ((size_t)b * NH + (n0 + ln)) * NC;

  // ---- pass 2: attn @ v, chunked through LDS ----
  v8f acc = {};
  for (int chunk = 0; chunk < 4; ++chunk) {
#pragma unroll 4
    for (int s2 = 0; s2 < 32; ++s2) {
      int jj = g + (s2 << 4);
      int j = (chunk << 9) + jj;
      float sc = sqv + skb[j];
      sc = sc > 0.f ? sc : NEG_SLOPE * sc;
      sc *= (float)mrow[j];
      sAttn[i][jj] = (h16)(__expf(sc - mi) * inv);
    }
    __syncthreads();
    const h16* arow = &sAttn[ln][0];
    const h16* brow = brow0 + (chunk << 9);
#pragma unroll
    for (int kk = 0; kk < 512; kk += 32) {
      v16h a = ld_a_frag(arow, kk, hi);
      v16h bf = *(const v16h*)(brow + kk + hi * 16);
      acc = wmma_f16(a, bf, acc);
    }
    __syncthreads();  // sAttn reused next chunk
  }

  // ---- stage weighted tile, project through out_w ----
#pragma unroll
  for (int r = 0; r < 8; ++r)
    sWh[r + hi * 8][n0 + ln] = (h16)acc[r];
  __syncthreads();

  v8f acc2 = {};
  const h16* arow2 = &sWh[ln][0];
  const h16* brow2 = outw_h + (size_t)(n0 + ln) * NH;
#pragma unroll
  for (int kk = 0; kk < NH; kk += 32) {
    v16h a = ld_a_frag(arow2, kk, hi);
    v16h bf = *(const v16h*)(brow2 + kk + hi * 16);
    acc2 = wmma_f16(a, bf, acc2);
  }
  const float bias = out_b[n0 + ln];
#pragma unroll
  for (int r = 0; r < 8; ++r)
    out[((size_t)b * NC + i0 + r + hi * 8) * ND + n0 + ln] = acc2[r] + bias;
}

// ---------------------------------------------------------------------
extern "C" void kernel_launch(void* const* d_in, const int* in_sizes, int n_in,
                              void* d_out, int out_size, void* d_ws,
                              size_t ws_size, hipStream_t stream) {
  (void)in_sizes; (void)n_in; (void)out_size; (void)ws_size;
  const float* features = (const float*)d_in[0];
  const int* mask       = (const int*)d_in[1];
  const float* w_w      = (const float*)d_in[2];
  const float* w_b      = (const float*)d_in[3];
  const float* score_w  = (const float*)d_in[4];
  const float* score_b  = (const float*)d_in[5];
  const float* out_w    = (const float*)d_in[6];
  const float* out_b    = (const float*)d_in[7];
  float* out = (float*)d_out;

  // workspace layout (~4.4 MB total)
  char* ws = (char*)d_ws;
  float* vq    = (float*)(ws + 0);                     // 512 B
  float* vk    = (float*)(ws + 512);                   // 512 B
  float* bqk   = (float*)(ws + 1024);                  // 8 B
  h16* wv_h    = (h16*)(ws + 4096);                    // 32 KB
  h16* outw_h  = (h16*)(ws + 4096 + 32768);            // 32 KB
  float* sq    = (float*)(ws + 69632);                 // 64 KB
  float* sk    = (float*)(ws + 135168);                // 64 KB
  h16* vT      = (h16*)(ws + 200704);                  // 4 MB, 32B aligned

  prep_kernel<<<1, 256, 0, stream>>>(w_w, w_b, score_w, score_b, out_w, vq, vk,
                                     bqk, wv_h, outw_h);
  qkv_kernel<<<NB * NC / 16, 256, 0, stream>>>(features, w_b, vq, vk, bqk, wv_h,
                                               sq, sk, vT);
  attn_kernel<<<NB * NC / 16, 256, 0, stream>>>(mask, sq, sk, vT, outw_h, out_b,
                                                out);
}
